// DepthNet_51599737094655
// MI455X (gfx1250) — matrix operand order repaired and checked by reading, not compiled, not run on hardware
//
#include <hip/hip_runtime.h>

// ---------------------------------------------------------------------------
// Problem constants (B=1, C=32, H=128, W=160, D=48, V=3)
// ---------------------------------------------------------------------------
#define Hc   128
#define Wc   160
#define Dc   48
#define Cc   32
#define HWc  (Hc * Wc)            // 20480
// Zero-padded variance volume: z in [0,50), y in [0,130), x in [0,162)
#define Zp   (Dc + 2)             // 50
#define Yp   (Hc + 2)             // 130
#define Xp   (Wc + 2)             // 162
#define NCHUNK 162                // 3(dz) * 3(dy) * 18(xx) K-chunks of 32 (=C)
#define PAD_ELEMS ((size_t)Zp * Yp * Xp * Cc)       // bf16 elements

typedef __attribute__((ext_vector_type(16))) __bf16 v16bf;
typedef __attribute__((ext_vector_type(8)))  float  v8f;

union FragBF { v16bf v; uint4 q[2]; unsigned short s[16]; };

__device__ __forceinline__ unsigned short f2bf(float f) {
    unsigned int u   = __float_as_uint(f);
    unsigned int lsb = (u >> 16) & 1u;
    return (unsigned short)((u + 0x7FFFu + lsb) >> 16);   // round-to-nearest-even
}

// ---------------------------------------------------------------------------
// Kernel 0: zero the padded variance volume (halo must be 0; interior is
// overwritten by warp_var_kernel). 67 MB @ 23.3 TB/s ~ 3 us.
// ---------------------------------------------------------------------------
__global__ void zero_kernel(uint4* __restrict__ p, unsigned long long n4) {
    unsigned long long i      = (unsigned long long)blockIdx.x * blockDim.x + threadIdx.x;
    unsigned long long stride = (unsigned long long)gridDim.x * blockDim.x;
    uint4 z = make_uint4(0, 0, 0, 0);
    for (; i < n4; i += stride) p[i] = z;
}

// ---------------------------------------------------------------------------
// Kernel 1: invert ref_proj, compute per-view rot(3x3)+trans(3) for views 1,2
// rt layout: view v-1 -> 12 floats: rot row-major [0..8], trans [9..11]
// ---------------------------------------------------------------------------
__global__ void prep_proj_kernel(const float* __restrict__ proj, float* __restrict__ rt) {
    if (threadIdx.x || blockIdx.x) return;
    float a[4][8];
    for (int r = 0; r < 4; ++r)
        for (int c = 0; c < 4; ++c) { a[r][c] = proj[r * 4 + c]; a[r][4 + c] = (r == c) ? 1.f : 0.f; }
    for (int col = 0; col < 4; ++col) {
        int piv = col; float best = fabsf(a[col][col]);
        for (int r = col + 1; r < 4; ++r) { float v = fabsf(a[r][col]); if (v > best) { best = v; piv = r; } }
        if (piv != col) for (int c = 0; c < 8; ++c) { float t = a[col][c]; a[col][c] = a[piv][c]; a[piv][c] = t; }
        float inv = 1.f / a[col][col];
        for (int c = 0; c < 8; ++c) a[col][c] *= inv;
        for (int r = 0; r < 4; ++r) {
            if (r == col) continue;
            float f = a[r][col];
            for (int c = 0; c < 8; ++c) a[r][c] -= f * a[col][c];
        }
    }
    for (int v = 1; v < 3; ++v) {
        const float* pv = proj + v * 16;
        float* o = rt + (v - 1) * 12;
        for (int r = 0; r < 3; ++r) {
            float m[4];
            for (int c = 0; c < 4; ++c) {
                float s = 0.f;
                for (int k = 0; k < 4; ++k) s += pv[r * 4 + k] * a[k][4 + c];
                m[c] = s;
            }
            o[r * 3 + 0] = m[0]; o[r * 3 + 1] = m[1]; o[r * 3 + 2] = m[2]; o[9 + r] = m[3];
        }
    }
}

// ---------------------------------------------------------------------------
// Kernel 2: bake shifted-weight B matrices in WMMA bf16 B-fragment layout.
// B[k=c, n] for chunk (dz,dy,xx): w[c,dz,dy,xx-n] if 0<=xx-n<=2 else 0.
// Per-lane layout: lane l holds n=l&15, k=(l>>4)*16+j for j=0..15, contiguous.
// ---------------------------------------------------------------------------
__global__ void bmat_kernel(const float* __restrict__ w, unsigned short* __restrict__ bmat) {
    int idx = blockIdx.x * blockDim.x + threadIdx.x;
    if (idx >= NCHUNK * 512) return;
    int chunk = idx >> 9;
    int l     = (idx >> 4) & 31;
    int j     = idx & 15;
    int c     = ((l >> 4) << 4) + j;        // k index == input channel
    int n     = l & 15;
    int dz    = chunk / 54;
    int dy    = (chunk / 18) % 3;
    int xx    = chunk % 18;
    int dx    = xx - n;
    float val = 0.f;
    if (dx >= 0 && dx <= 2) val = w[((c * 3 + dz) * 3 + dy) * 3 + dx];
    bmat[idx] = f2bf(val);
}

// ---------------------------------------------------------------------------
// Kernel 3: homography warp + 3-view variance -> padded bf16 [z][y][x][c]
// ---------------------------------------------------------------------------
__global__ void warp_var_kernel(const float* __restrict__ f0,
                                const float* __restrict__ f1,
                                const float* __restrict__ f2,
                                const float* __restrict__ rt,
                                const float* __restrict__ depthv,
                                unsigned short* __restrict__ var) {
    int idx = blockIdx.x * blockDim.x + threadIdx.x;
    if (idx >= Dc * HWc) return;
    int d = idx / HWc;
    int p = idx - d * HWc;
    int y = p / Wc;
    int x = p - y * Wc;
    float depth = depthv[d];

    int   off[2][4];
    float wgt[2][4];
#pragma unroll
    for (int v = 0; v < 2; ++v) {
        const float* r = rt + v * 12;
        float fx = (float)x, fy = (float)y;
        float px = (r[0] * fx + r[1] * fy + r[2]) * depth + r[9];
        float py = (r[3] * fx + r[4] * fy + r[5]) * depth + r[10];
        float pz = (r[6] * fx + r[7] * fy + r[8]) * depth + r[11];
        float gx = px / pz, gy = py / pz;
        float x0f = floorf(gx), y0f = floorf(gy);
        float wx = gx - x0f, wy = gy - y0f;
        float xs[2]  = { x0f, x0f + 1.f };
        float ys[2]  = { y0f, y0f + 1.f };
        float wxs[2] = { 1.f - wx, wx };
        float wys[2] = { 1.f - wy, wy };
        int k = 0;
#pragma unroll
        for (int cy = 0; cy < 2; ++cy)
#pragma unroll
            for (int cx = 0; cx < 2; ++cx, ++k) {
                float xi = xs[cx], yi = ys[cy];
                bool valid = (xi >= 0.f) && (xi <= (float)(Wc - 1)) &&
                             (yi >= 0.f) && (yi <= (float)(Hc - 1));
                int xc = min(max((int)xi, 0), Wc - 1);
                int yc = min(max((int)yi, 0), Hc - 1);
                off[v][k] = yc * Wc + xc;
                wgt[v][k] = valid ? (wys[cy] * wxs[cx]) : 0.f;
            }
    }

    // write into the interior of the padded volume
    unsigned short* out = var + (((size_t)(d + 1) * Yp + (y + 1)) * Xp + (x + 1)) * Cc;
    for (int c = 0; c < Cc; ++c) {
        const int cb = c * HWc;
        float refv = f0[cb + p];
        float s = refv, sq = refv * refv;
#pragma unroll
        for (int v = 0; v < 2; ++v) {
            const float* fc = (v == 0 ? f1 : f2) + cb;
            float wv = fc[off[v][0]] * wgt[v][0] + fc[off[v][1]] * wgt[v][1] +
                       fc[off[v][2]] * wgt[v][2] + fc[off[v][3]] * wgt[v][3];
            s += wv; sq += wv * wv;
        }
        float vf = sq * (1.f / 3.f) - (s * (1.f / 3.f)) * (s * (1.f / 3.f));
        out[c] = f2bf(vf);
    }
}

// ---------------------------------------------------------------------------
// Kernel 4: 3x3x3 conv (1 output channel) as implicit GEMM on bf16 WMMA.
// One wave per 16x16(y,x) x 1(d) tile; 162 x v_wmma_f32_16x16x32_bf16.
// A: unconditional 2x16B global loads from the padded volume (no predication).
// B: staged per-dz (54 KB) into LDS once per block (all 8 waves share d since
//    8 | 80 tiles-per-d), read back with ds_load_b128.
// (b_reg dropped: softmax is shift-invariant.)
// ---------------------------------------------------------------------------
__global__ void conv_wmma_kernel(const unsigned short* __restrict__ var,
                                 const unsigned short* __restrict__ bmat,
                                 float* __restrict__ cost) {
    __shared__ unsigned short sB[54 * 512];        // 55,296 B: B(dz) fragments
    uint4*       sq = (uint4*)sB;                  // 3456 uint4
    const uint4* bq = (const uint4*)bmat;
    const uint4* vq = (const uint4*)var;

    int tid   = threadIdx.x;
    int lane  = tid & 31;
    int wave  = tid >> 5;
    int tile  = blockIdx.x * 8 + wave;             // 48*8*10 = 3840 tiles
    int d     = tile / 80;                         // uniform across the block
    int r     = tile - d * 80;
    int ty    = r / 10;
    int tx    = r - ty * 10;
    int y0    = ty * 16, x0 = tx * 16;
    int m     = lane & 15;
    int khalf = lane >> 4;

    v8f acc = {};
    for (int dz = 0; dz < 3; ++dz) {
        __syncthreads();
        for (int i = tid; i < 54 * 64; i += 256)   // cooperative B(dz) -> LDS
            sq[i] = bq[dz * 54 * 64 + i];
        __syncthreads();

        int zrow = d + dz;                         // padded z index
#pragma unroll
        for (int dy = 0; dy < 3; ++dy) {
            int    yrow = y0 + m + dy;             // padded y index
            size_t rq   = ((((size_t)zrow * Yp + yrow) * Xp + x0) * Cc + khalf * 16) >> 3;
            int    clb  = (dy * 18) * 32;          // LDS chunk base (in 32-lane units)
            for (int xx = 0; xx < 18; ++xx) {
                FragBF a, b;
                a.q[0] = vq[rq];                   // c = khalf*16 .. +7
                a.q[1] = vq[rq + 1];               // c = khalf*16+8 .. +15
                rq += 4;                           // next x: +32 bf16 elements
                size_t bb = ((size_t)(clb + xx * 32 + lane)) * 2;
                b.q[0] = sq[bb];
                b.q[1] = sq[bb + 1];
                acc = __builtin_amdgcn_wmma_f32_16x16x32_bf16(
                          false, a.v, false, b.v, (short)0, acc, false, false);
            }
        }
    }

    // D layout: lane -> n = lane&15, rows M = khalf*8 + rr
    int n = lane & 15;
#pragma unroll
    for (int rr = 0; rr < 8; ++rr) {
        int yo = y0 + khalf * 8 + rr;
        cost[(size_t)d * HWc + yo * Wc + x0 + n] = acc[rr];
    }
}

// ---------------------------------------------------------------------------
// Kernel 5: softmax over D -> expected depth + max-prob confidence
// ---------------------------------------------------------------------------
__global__ void softmax_kernel(const float* __restrict__ cost,
                               const float* __restrict__ depthv,
                               float* __restrict__ out) {
    int p = blockIdx.x * blockDim.x + threadIdx.x;
    if (p >= HWc) return;
    float mx = -3.4e38f;
    for (int d = 0; d < Dc; ++d) mx = fmaxf(mx, cost[(size_t)d * HWc + p]);
    float s = 0.f, ds = 0.f, pm = 0.f;
    for (int d = 0; d < Dc; ++d) {
        float e = __expf(cost[(size_t)d * HWc + p] - mx);
        s  += e;
        ds += e * depthv[d];
        pm  = fmaxf(pm, e);
    }
    out[p]       = ds / s;   // depth
    out[HWc + p] = pm / s;   // photometric confidence
}

// ---------------------------------------------------------------------------
extern "C" void kernel_launch(void* const* d_in, const int* in_sizes, int n_in,
                              void* d_out, int out_size, void* d_ws, size_t ws_size,
                              hipStream_t stream) {
    (void)in_sizes; (void)n_in; (void)out_size; (void)ws_size;
    const float* feat0  = (const float*)d_in[0];
    const float* feat1  = (const float*)d_in[1];
    const float* feat2  = (const float*)d_in[2];
    const float* proj   = (const float*)d_in[3];   // (1,3,4,4)
    const float* depthv = (const float*)d_in[4];   // (1,48)
    const float* w_reg  = (const float*)d_in[5];   // (1,32,3,3,3)
    // d_in[6] = b_reg (shift-invariant under softmax, ignored), d_in[7] = num_depth

    char* ws = (char*)d_ws;
    size_t off = 0;
    unsigned short* var  = (unsigned short*)(ws + off); off += PAD_ELEMS * 2;             // 67.4 MB
    unsigned short* bmat = (unsigned short*)(ws + off); off += (size_t)NCHUNK * 512 * 2;  // 166 KB
    float*          rt   = (float*)(ws + off);          off += 24 * sizeof(float);
    off = (off + 255) & ~(size_t)255;
    float*          cost = (float*)(ws + off);          // 3.9 MB

    unsigned long long n4 = (PAD_ELEMS * 2) / 16;       // uint4 count
    hipLaunchKernelGGL(zero_kernel, dim3(2048), dim3(256), 0, stream, (uint4*)var, n4);
    hipLaunchKernelGGL(prep_proj_kernel, dim3(1), dim3(32), 0, stream, proj, rt);
    hipLaunchKernelGGL(bmat_kernel, dim3((NCHUNK * 512 + 255) / 256), dim3(256), 0, stream,
                       w_reg, bmat);
    hipLaunchKernelGGL(warp_var_kernel, dim3((Dc * HWc + 255) / 256), dim3(256), 0, stream,
                       feat0, feat1, feat2, rt, depthv, var);
    hipLaunchKernelGGL(conv_wmma_kernel, dim3(480), dim3(256), 0, stream, var, bmat, cost);
    hipLaunchKernelGGL(softmax_kernel, dim3((HWc + 255) / 256), dim3(256), 0, stream,
                       cost, depthv, (float*)d_out);
}
